// XPainnMessage_26963804684388
// MI455X (gfx1250) — compile-verified, matrix operand order checked by reading, not run on hardware
//
#include <hip/hip_runtime.h>
#include <hip/hip_bf16.h>
#include <math.h>

// ---------------------------------------------------------------------------
// XPainnMessage fused pipeline for MI455X (gfx1250, wave32, WMMA).
// All matmuls run on v_wmma_f32_16x16x32_f16 (f16 in, f32 accumulate).
// Edge intermediates (filter_weight/filter_out, 230MB each if materialized)
// stay in LDS; scatter-add via global f32 atomics (L2-resident targets).
// Streaming inputs (rsh 192MB, rbf 8MB) use non-temporal loads so the
// L2-resident gather set (inv_out 23MB + ne 19MB + outputs 25MB) survives.
// ---------------------------------------------------------------------------

typedef __attribute__((ext_vector_type(16))) _Float16 v16h;
typedef __attribute__((ext_vector_type(8)))  _Float16 v8h;
typedef __attribute__((ext_vector_type(8)))  float    v8f;

#define N_NODES   10000
#define N_EDGES   100000
#define NODE_DIM  128
#define EQUI_DIM  480
#define HIDDEN    576
#define NUM_BASIS 20
#define NUM_IRREPS 224

// Build a v16h WMMA fragment from two contiguous 8-half (16B) chunks.
// 16-bit A 16x32 layout (ISA 7.12.2): lane half h=lane/16 selects K offsets
//   halves [0..7] = K[8h .. 8h+7], halves [8..15] = K[16+8h .. 23+8h].
// B (32x16) is the mirrored layout with N = lane%16; with B stored
// column-major (Wt[n][k]) the same two 16B loads produce the B fragment.
__device__ inline v16h make_frag(const _Float16* lo, const _Float16* hi) {
    v8h a = *(const v8h*)lo;
    v8h b = *(const v8h*)hi;
    return __builtin_shufflevector(a, b, 0,1,2,3,4,5,6,7,8,9,10,11,12,13,14,15);
}

__device__ inline v8f wmma_f16(v16h a, v16h b, v8f c) {
    return __builtin_amdgcn_wmma_f32_16x16x32_f16(false, a, false, b,
                                                  (short)0, c, false, false);
}

// SiLU with hardware reciprocal: v_exp_f32 + v_rcp_f32 (result feeds an f16
// store, so ~1ulp rcp is more than enough precision).
__device__ inline float silu_fast(float x) {
    return x * __builtin_amdgcn_rcpf(1.f + __expf(-x));
}

// ---------------------------------------------------------------------------
// Kernel 0: convert weights to transposed f16 layouts for B-fragment loads.
//   W1t [128][128] (n-major), W2t [576][128], rbfWt [576][32] (K padded 20->32)
// ---------------------------------------------------------------------------
__global__ void convert_weights_kernel(const float* __restrict__ W1,
                                       const float* __restrict__ W2,
                                       const float* __restrict__ rbfW,
                                       _Float16* __restrict__ W1t,
                                       _Float16* __restrict__ W2t,
                                       _Float16* __restrict__ rbfWt) {
    int tid = blockIdx.x * blockDim.x + threadIdx.x;
    int stride = gridDim.x * blockDim.x;
    for (int i = tid; i < NODE_DIM * NODE_DIM; i += stride) {
        int n = i / NODE_DIM, k = i % NODE_DIM;
        W1t[i] = (_Float16)W1[k * NODE_DIM + n];
    }
    for (int i = tid; i < HIDDEN * NODE_DIM; i += stride) {
        int n = i / NODE_DIM, k = i % NODE_DIM;
        W2t[i] = (_Float16)W2[k * HIDDEN + n];
    }
    for (int i = tid; i < HIDDEN * 32; i += stride) {
        int n = i / 32, k = i % 32;
        rbfWt[i] = (k < NUM_BASIS) ? (_Float16)rbfW[k * HIDDEN + n] : (_Float16)0.0f;
    }
}

// ---------------------------------------------------------------------------
// Kernel 1: o3_layer_norm -> ne workspace. One wave per node.
// ---------------------------------------------------------------------------
__global__ __launch_bounds__(256) void o3_ln_kernel(const float* __restrict__ x,
                                                    const float* __restrict__ w,
                                                    const float* __restrict__ b,
                                                    float* __restrict__ ne) {
    int wave = threadIdx.x >> 5;
    int lane = threadIdx.x & 31;
    int node = blockIdx.x * 8 + wave;
    if (node >= N_NODES) return;
    const float* xr = x + (size_t)node * EQUI_DIM;
    float* outr = ne + (size_t)node * EQUI_DIM;

    // scalar part: 128 dims, standard layer_norm
    float vals[4], s = 0.f, ss = 0.f;
    for (int j = 0; j < 4; j++) {
        float v = xr[lane + 32 * j];
        vals[j] = v; s += v; ss += v * v;
    }
    for (int d = 1; d < 32; d <<= 1) { s += __shfl_xor(s, d, 32); ss += __shfl_xor(ss, d, 32); }
    float mu = s * (1.f / 128.f);
    float var = ss * (1.f / 128.f) - mu * mu;
    float rinv = rsqrtf(var + 1e-5f);
    for (int j = 0; j < 4; j++) {
        int c = lane + 32 * j;
        outr[c] = (vals[j] - mu) * rinv * w[c] + b[c];
    }
    // l=1 part: 64 vectors of 3 (dims 128..319)
    float v1[6]; float sq = 0.f;
    for (int j = 0; j < 6; j++) {
        float v = xr[128 + lane + 32 * j];
        v1[j] = v; sq += v * v;
    }
    for (int d = 1; d < 32; d <<= 1) sq += __shfl_xor(sq, d, 32);
    float r1 = rsqrtf(sq * (1.f / 64.f) + 1e-5f);
    for (int j = 0; j < 6; j++) {
        int idx = lane + 32 * j;
        outr[128 + idx] = v1[j] * r1 * w[128 + idx / 3];
    }
    // l=2 part: 32 vectors of 5 (dims 320..479)
    float v2[5]; sq = 0.f;
    for (int j = 0; j < 5; j++) {
        float v = xr[320 + lane + 32 * j];
        v2[j] = v; sq += v * v;
    }
    for (int d = 1; d < 32; d <<= 1) sq += __shfl_xor(sq, d, 32);
    float r2 = rsqrtf(sq * (1.f / 32.f) + 1e-5f);
    for (int j = 0; j < 5; j++) {
        int idx = lane + 32 * j;
        outr[320 + idx] = v2[j] * r2 * w[192 + idx / 5];
    }
}

// ---------------------------------------------------------------------------
// Kernel 2: per-node MLP.  16 nodes/block, 256 threads (8 waves).
//   ns = layer_norm(node_invariant) ; h = silu(ns@W1+b1) ; inv_out = h@W2+b2
// GEMM1: 8 N-tiles of 16 (one per wave), K=128 -> 4 WMMA steps.
// GEMM2: 36 N-tiles round-robin over waves, K=128 -> 4 WMMA steps.
// ---------------------------------------------------------------------------
__global__ __launch_bounds__(256) void node_mlp_kernel(const float* __restrict__ ninv,
                                                       const float* __restrict__ lnw,
                                                       const float* __restrict__ lnb,
                                                       const _Float16* __restrict__ W1t,
                                                       const float* __restrict__ b1,
                                                       const _Float16* __restrict__ W2t,
                                                       const float* __restrict__ b2,
                                                       float* __restrict__ inv_out) {
    __shared__ alignas(16) _Float16 As[16][136];  // ns tile, f16, 272B row (16B multiple)
    __shared__ alignas(16) _Float16 Hs[16][136];  // silu hidden tile
    const int tid = threadIdx.x;
    const int node0 = blockIdx.x * 16;

    // ---- LayerNorm: 16 threads per row, 8 cols each ----
    {
        int row = tid >> 4;
        int c16 = tid & 15;
        const float* xr = ninv + (size_t)(node0 + row) * NODE_DIM;
        float vals[8], s = 0.f, ss = 0.f;
        for (int j = 0; j < 8; j++) {
            float v = xr[c16 + 16 * j];
            vals[j] = v; s += v; ss += v * v;
        }
        for (int d = 1; d < 16; d <<= 1) { s += __shfl_xor(s, d, 32); ss += __shfl_xor(ss, d, 32); }
        float mu = s * (1.f / 128.f);
        float var = ss * (1.f / 128.f) - mu * mu;
        float rinv = rsqrtf(var + 1e-5f);
        for (int j = 0; j < 8; j++) {
            int col = c16 + 16 * j;
            As[row][col] = (_Float16)((vals[j] - mu) * rinv * lnw[col] + lnb[col]);
        }
    }
    __syncthreads();

    const int wave = tid >> 5;
    const int lane = tid & 31;
    const int m = lane & 15;       // M (A) / N (B,C) index within tile
    const int h = lane >> 4;       // lane half selects K sub-pattern / C row offset

    // ---- GEMM1 + SiLU -> Hs ----
    {
        v8f acc = {};
        const _Float16* bt = W1t + (size_t)(wave * 16 + m) * NODE_DIM;
        const _Float16* ar = &As[m][0];
        for (int kk = 0; kk < 4; kk++) {
            v16h a = make_frag(ar + kk * 32 + 8 * h, ar + kk * 32 + 16 + 8 * h);
            v16h b = make_frag(bt + kk * 32 + 8 * h, bt + kk * 32 + 16 + 8 * h);
            acc = wmma_f16(a, b, acc);
        }
        int col = wave * 16 + m;
        float bias = b1[col];
        for (int r = 0; r < 8; r++) {
            Hs[r + 8 * h][col] = (_Float16)silu_fast(acc[r] + bias);
        }
    }
    __syncthreads();

    // ---- GEMM2 -> inv_out (576 cols = 36 tiles) ----
    for (int t = wave; t < 36; t += 8) {
        v8f acc = {};
        const _Float16* bt = W2t + (size_t)(t * 16 + m) * NODE_DIM;
        const _Float16* ar = &Hs[m][0];
        for (int kk = 0; kk < 4; kk++) {
            v16h a = make_frag(ar + kk * 32 + 8 * h, ar + kk * 32 + 16 + 8 * h);
            v16h b = make_frag(bt + kk * 32 + 8 * h, bt + kk * 32 + 16 + 8 * h);
            acc = wmma_f16(a, b, acc);
        }
        int col = t * 16 + m;
        float bias = b2[col];
        for (int r = 0; r < 8; r++) {
            inv_out[(size_t)(node0 + r + 8 * h) * HIDDEN + col] = acc[r] + bias;
        }
    }
}

// ---------------------------------------------------------------------------
// Kernel 3: per-edge fused filter + message + scatter.  16 edges/block.
//   fw  = (rbf @ rbf_W + rbf_b) * fcut          (K=20 padded to 32: ONE wmma/tile)
//   fo  = inv_out[nbr] * fw                     (gather fused in epilogue, in LDS)
//   invt scatter: atomicAdd(out_invt[center], fo[:,448:576])
//   equi scatter: atomicAdd(out_equi[center], ne[nbr]*gate_s + rsh*gate_e)
// ---------------------------------------------------------------------------
__global__ __launch_bounds__(256) void edge_kernel(const float* __restrict__ rbf,
                                                   const float* __restrict__ fcut,
                                                   const float* __restrict__ rsh,
                                                   const int* __restrict__ edge_index,
                                                   const _Float16* __restrict__ rbfWt,
                                                   const float* __restrict__ rbf_b,
                                                   const float* __restrict__ inv_out,
                                                   const float* __restrict__ ne,
                                                   float* __restrict__ out_invt,
                                                   float* __restrict__ out_equi) {
    __shared__ alignas(16) _Float16 Ars[16][40];       // rbf tile f16, K padded to 32
    __shared__ float fo[16][HIDDEN + 4];               // filter_out tile (37 KB)
    __shared__ int   s_ctr[16], s_nbr[16];
    __shared__ float s_fcut[16];

    const int tid = threadIdx.x;
    const int edge0 = blockIdx.x * 16;

    // rbf is read exactly once across the dispatch: non-temporal load so it
    // doesn't evict the L2-resident inv_out/ne gather set.
    for (int i = tid; i < 16 * 32; i += 256) {
        int r = i >> 5, k = i & 31;
        float v = (k < NUM_BASIS)
                    ? __builtin_nontemporal_load(&rbf[(size_t)(edge0 + r) * NUM_BASIS + k])
                    : 0.0f;
        Ars[r][k] = (_Float16)v;
    }
    if (tid < 16) {
        int e = edge0 + tid;
        s_ctr[tid]  = edge_index[e];                 // edge_index[0][e]
        s_nbr[tid]  = edge_index[N_EDGES + e];       // edge_index[1][e]
        s_fcut[tid] = fcut[e];
    }
    __syncthreads();

    const int wave = tid >> 5;
    const int lane = tid & 31;
    const int m = lane & 15;
    const int h = lane >> 4;

    // A fragment (same for all 36 N-tiles): rbf row m, K=0..31
    v16h a = make_frag(&Ars[m][8 * h], &Ars[m][16 + 8 * h]);

    for (int t = wave; t < 36; t += 8) {
        const _Float16* bt = rbfWt + (size_t)(t * 16 + m) * 32;
        v16h b = make_frag(bt + 8 * h, bt + 16 + 8 * h);
        v8f acc = {};
        acc = wmma_f16(a, b, acc);
        int col = t * 16 + m;
        float bias = rbf_b[col];
        for (int r = 0; r < 8; r++) {
            int rm = r + 8 * h;
            float fw = (acc[r] + bias) * s_fcut[rm];
            fo[rm][col] = fw * inv_out[(size_t)s_nbr[rm] * HIDDEN + col];
        }
    }
    __syncthreads();

    // ---- message + scatter: 16 threads per edge ----
    const int le  = tid >> 4;
    const int t16 = tid & 15;
    const int ctr = s_ctr[le];
    const int nbr = s_nbr[le];
    const int e   = edge0 + le;

    // invariant message: filter_out[:, 448:576]
    for (int j = 0; j < 8; j++) {
        int d = t16 + 16 * j;
        atomicAdd(&out_invt[(size_t)ctr * NODE_DIM + d], fo[le][2 * NUM_IRREPS + d]);
    }
    // equivariant message: ne[nbr]*expand(gate_state) + rsh*expand(gate_edge)
    // rsh is a 192MB single-use stream: non-temporal loads keep it out of L2.
    for (int j = 0; j < 30; j++) {
        int d = t16 + 16 * j;   // 0..479
        int g = (d < 128) ? d : ((d < 320) ? 128 + (d - 128) / 3
                                           : 192 + (d - 320) / 5);
        float rv = __builtin_nontemporal_load(&rsh[(size_t)e * EQUI_DIM + d]);
        float me = ne[(size_t)nbr * EQUI_DIM + d] * fo[le][g]
                 + rv * fo[le][NUM_IRREPS + g];
        atomicAdd(&out_equi[(size_t)ctr * EQUI_DIM + d], me);
    }
}

// ---------------------------------------------------------------------------
extern "C" void kernel_launch(void* const* d_in, const int* in_sizes, int n_in,
                              void* d_out, int out_size, void* d_ws, size_t ws_size,
                              hipStream_t stream) {
    const float* node_invariant   = (const float*)d_in[0];
    const float* node_equivariant = (const float*)d_in[1];
    const float* rbf              = (const float*)d_in[2];
    const float* fcut             = (const float*)d_in[3];
    const float* rsh              = (const float*)d_in[4];
    const int*   edge_index       = (const int*)d_in[5];
    const float* ln_w             = (const float*)d_in[6];
    const float* ln_b             = (const float*)d_in[7];
    const float* o3_w             = (const float*)d_in[8];
    const float* o3_b             = (const float*)d_in[9];
    const float* W1               = (const float*)d_in[10];
    const float* b1               = (const float*)d_in[11];
    const float* W2               = (const float*)d_in[12];
    const float* b2               = (const float*)d_in[13];
    const float* rbf_W            = (const float*)d_in[14];
    const float* rbf_b            = (const float*)d_in[15];

    float* out_invt = (float*)d_out;
    float* out_equi = out_invt + (size_t)N_NODES * NODE_DIM;

    // workspace layout (all slices 16B aligned): ~42.5 MB
    float*    inv_out = (float*)d_ws;                                   // 23.04 MB
    float*    ne      = inv_out + (size_t)N_NODES * HIDDEN;             // 19.20 MB
    _Float16* W1t     = (_Float16*)(ne + (size_t)N_NODES * EQUI_DIM);   // 32 KB
    _Float16* W2t     = W1t + NODE_DIM * NODE_DIM;                      // 144 KB
    _Float16* rbfWt   = W2t + NODE_DIM * HIDDEN;                        // 36 KB

    convert_weights_kernel<<<64, 256, 0, stream>>>(W1, W2, rbf_W, W1t, W2t, rbfWt);
    o3_ln_kernel<<<N_NODES / 8, 256, 0, stream>>>(node_equivariant, o3_w, o3_b, ne);
    node_mlp_kernel<<<N_NODES / 16, 256, 0, stream>>>(node_invariant, ln_w, ln_b,
                                                      W1t, b1, W2t, b2, inv_out);
    // initialize outputs with the residual inputs (scatter targets)
    hipMemcpyAsync(out_invt, node_invariant, (size_t)N_NODES * NODE_DIM * sizeof(float),
                   hipMemcpyDeviceToDevice, stream);
    hipMemcpyAsync(out_equi, node_equivariant, (size_t)N_NODES * EQUI_DIM * sizeof(float),
                   hipMemcpyDeviceToDevice, stream);
    edge_kernel<<<N_EDGES / 16, 256, 0, stream>>>(rbf, fcut, rsh, edge_index, rbfWt,
                                                  rbf_b, inv_out, ne, out_invt, out_equi);
}
